// SrpPhat_26534307955222
// MI455X (gfx1250) — matrix operand order repaired and checked by reading, not compiled, not run on hardware
//
#include <hip/hip_runtime.h>
#include <hip/hip_bf16.h>
#include <math.h>

// ---------------- problem constants (match reference) ----------------
#define N_MICS    8
#define P_PAIRS   36          // n_mics*(n_mics+1)/2
#define F_BINS    513
#define M_TOTAL   1000        // B*T = 4*250
#define D_DOAS    2562        // icosphere level 4
#define K_REAL    (F_BINS * P_PAIRS)          // 18468 (per re/im half)
#define KP_HALF   18496                       // pad to multiple of 32
#define KP        (2 * KP_HALF)               // 36992 = re||im concat
#define MP        1024                        // M padded: 16 tiles of 64
#define NP        2688                        // D padded: 21 tiles of 128
#define KCHUNK    128                         // K halves staged in LDS per chunk
#define STRIDE_A  (KCHUNK + 8)                // 136 halves: bank-conflict-free rows
#define OMEGA_SCALE 0.006135923151542565f     // 2*pi/1024

// workspace layout (all 256B aligned: row pitch 73984B = 289*256)
#define XH_BYTES  ((size_t)MP * KP * 2)       //  75,759,616 B
#define WH_BYTES  ((size_t)NP * KP * 2)       // 198,868,992 B

typedef __attribute__((ext_vector_type(16))) _Float16 v16h;
typedef __attribute__((ext_vector_type(8)))  _Float16 v8h;
typedef __attribute__((ext_vector_type(8)))  float    v8f;
typedef __attribute__((ext_vector_type(4)))  int      v4i;

#if __has_builtin(__builtin_amdgcn_global_load_async_to_lds_b128) && \
    __has_builtin(__builtin_amdgcn_s_wait_asynccnt)
#define HAVE_ASYNC_LDS 1
// builtin signature: (v4i AS(1)*, v4i AS(3)*, int offset, int cpol)
typedef __attribute__((address_space(1))) v4i* gv4i_ptr;
typedef __attribute__((address_space(3))) v4i* sv4i_ptr;
#endif

// triu_indices(8)
__device__ __constant__ int c_i0[P_PAIRS] = {
  0,0,0,0,0,0,0,0, 1,1,1,1,1,1,1, 2,2,2,2,2,2, 3,3,3,3,3, 4,4,4,4, 5,5,5, 6,6, 7};
__device__ __constant__ int c_i1[P_PAIRS] = {
  0,1,2,3,4,5,6,7, 1,2,3,4,5,6,7, 2,3,4,5,6,7, 3,4,5,6,7, 4,5,6,7, 5,6,7, 6,7, 7};

// monotonic float -> u32 key (order preserving for all finite floats)
__device__ __forceinline__ unsigned int fkey(float f) {
  unsigned int u = __float_as_uint(f);
  return (u & 0x80000000u) ? ~u : (u | 0x80000000u);
}

// ---------------- kernel 0: init argmax buffer ----------------
__global__ void srp_init_best(unsigned long long* __restrict__ best) {
  int m = blockIdx.x * blockDim.x + threadIdx.x;
  if (m < M_TOTAL) best[m] = 0ull;
}

// ---------------- kernel 1: PHAT-normalize X -> f16, pad ----------------
__global__ void srp_pack_x(const float* __restrict__ XXs, _Float16* __restrict__ Xh) {
  size_t idx = (size_t)blockIdx.x * blockDim.x + threadIdx.x;
  if (idx >= (size_t)MP * KP_HALF) return;
  int m = (int)(idx / KP_HALF);
  int k = (int)(idx % KP_HALF);
  _Float16 re_h = (_Float16)0.0f, im_h = (_Float16)0.0f;
  if (m < M_TOTAL && k < K_REAL) {
    int f = k / P_PAIRS;
    int p = k - f * P_PAIRS;
    size_t base = ((size_t)m * F_BINS + f) * (2 * P_PAIRS) + p;   // (B,T,F,2,P)
    float re = XXs[base];
    float im = XXs[base + P_PAIRS];
    float inv = 1.0f / (sqrtf(re * re + im * im) + 1e-20f);
    re_h = (_Float16)(re * inv);
    im_h = (_Float16)(im * inv);
  }
  size_t row = (size_t)m * KP;
  Xh[row + k]           = re_h;
  Xh[row + KP_HALF + k] = im_h;
}

// ---------------- kernel 2: steering weights W -> f16, pad ----------------
__global__ void srp_build_w(const float* __restrict__ taus, _Float16* __restrict__ Wh) {
  size_t idx = (size_t)blockIdx.x * blockDim.x + threadIdx.x;
  if (idx >= (size_t)NP * KP_HALF) return;
  int d = (int)(idx / KP_HALF);
  int k = (int)(idx % KP_HALF);
  _Float16 wr_h = (_Float16)0.0f, wi_h = (_Float16)0.0f;
  if (d < D_DOAS && k < K_REAL) {
    int f = k / P_PAIRS;
    int p = k - f * P_PAIRS;
    float dt    = taus[d * N_MICS + c_i0[p]] - taus[d * N_MICS + c_i1[p]];
    float delta = (OMEGA_SCALE * (float)f) * dt;
    float s, c;
    __sincosf(delta, &s, &c);
    wr_h = (_Float16)c;
    wi_h = (_Float16)s;
  }
  size_t row = (size_t)d * KP;
  Wh[row + k]           = wr_h;
  Wh[row + KP_HALF + k] = wi_h;
}

// stage one 64 x KCHUNK A-tile chunk into an LDS buffer (8 x 16B per thread)
__device__ __forceinline__ void stage_a(const _Float16* __restrict__ Xh,
                                        _Float16* lbuf, int mbase, int k0, int tid) {
  #pragma unroll
  for (int i = 0; i < 8; ++i) {
    int u   = i * 128 + tid;
    int row = u >> 4;
    int kk  = (u & 15) * 8;
    const _Float16* g = Xh + (size_t)(mbase + row) * KP + k0 + kk;
    _Float16* l = lbuf + row * STRIDE_A + kk;
#ifdef HAVE_ASYNC_LDS
    __builtin_amdgcn_global_load_async_to_lds_b128(
        (gv4i_ptr)g, (sv4i_ptr)l, /*offset=*/0, /*cpol=*/0);
#else
    *(v8h*)l = *(const v8h*)g;
#endif
  }
}

// ---------------- kernel 3: WMMA GEMM + fused argmax ----------------
// Block = 128 threads = 4 waves; tile 64(M) x 128(N); wave w owns N columns
// [w*32, w*32+32): 4 M-frags x 2 N-frags = 8 f32 accumulators.
// A tile is double-buffered in LDS via async global->LDS copies: chunk i+1 is
// issued before computing chunk i; s_wait_asynccnt(8) retires only the current
// buffer's 8 in-order copies. Fragments read with ds_load_b128 (padded stride
// -> conflict-free banks).
__global__ __launch_bounds__(128) void srp_gemm(const _Float16* __restrict__ Xh,
                                                const _Float16* __restrict__ Wh,
                                                unsigned long long* __restrict__ best) {
  __shared__ _Float16 ldsA[2][64 * STRIDE_A];   // 2 x 17KB

  const int tid   = threadIdx.x;
  const int lane  = tid & 31;
  const int wv    = tid >> 5;                // 0..3
  const int half  = lane >> 4;               // 0 | 1
  const int l16   = lane & 15;
  const int mbase = blockIdx.x * 64;
  const int nbase = blockIdx.y * 128 + wv * 32;

  // B fragment sources (32B contiguous per lane; lanes 0-15 K[0:16), 16-31 K[16:32))
  const _Float16* wrow0 = Wh + (size_t)(nbase + l16)      * KP + half * 16;
  const _Float16* wrow1 = Wh + (size_t)(nbase + 16 + l16) * KP + half * 16;

  v8f acc[4][2];
  #pragma unroll
  for (int mf = 0; mf < 4; ++mf) { acc[mf][0] = (v8f){}; acc[mf][1] = (v8f){}; }

  stage_a(Xh, &ldsA[0][0], mbase, 0, tid);     // prologue: chunk 0 -> buffer 0

  int buf = 0;
  for (int k0 = 0; k0 < KP; k0 += KCHUNK) {
    // issue next chunk into the other buffer (its readers finished at the
    // barrier that ended the previous iteration)
    if (k0 + KCHUNK < KP) {
      stage_a(Xh, &ldsA[buf ^ 1][0], mbase, k0 + KCHUNK, tid);
#ifdef HAVE_ASYNC_LDS
      __builtin_amdgcn_s_wait_asynccnt(8);     // current buffer's copies done
#endif
    } else {
#ifdef HAVE_ASYNC_LDS
      __builtin_amdgcn_s_wait_asynccnt(0);
#endif
    }
    __syncthreads();                            // current buffer visible to all

    const _Float16* lbuf = &ldsA[buf][0];
    #pragma unroll
    for (int kk = 0; kk < KCHUNK; kk += 32) {
      __builtin_prefetch(wrow0 + k0 + kk + 512, 0, 0);   // GL2, 4 chunks ahead
      __builtin_prefetch(wrow1 + k0 + kk + 512, 0, 0);

      v16h b0 = *(const v16h*)(wrow0 + k0 + kk);
      v16h b1 = *(const v16h*)(wrow1 + k0 + kk);

      // preload all 4 A fragments, then issue all 8 WMMAs (one dscnt wait)
      v16h af[4];
      #pragma unroll
      for (int mf = 0; mf < 4; ++mf) {
        // A 16x32 f16 lane layout: halves [0:16) at K+{0|16} (contiguous, matches B)
        const _Float16* lp = lbuf + (mf * 16 + l16) * STRIDE_A + kk + half * 16;
        v8h lo = *(const v8h*)lp;
        v8h hi = *(const v8h*)(lp + 8);
        af[mf] = __builtin_shufflevector(lo, hi,
                   0,1,2,3,4,5,6,7,8,9,10,11,12,13,14,15);
      }
      #pragma unroll
      for (int mf = 0; mf < 4; ++mf) {
        acc[mf][0] = __builtin_amdgcn_wmma_f32_16x16x32_f16(
                         false, af[mf], false, b0, (short)0, acc[mf][0], false, false);
        acc[mf][1] = __builtin_amdgcn_wmma_f32_16x16x32_f16(
                         false, af[mf], false, b1, (short)0, acc[mf][1], false, false);
      }
    }
    __syncthreads();                            // readers done -> buffer reusable
    buf ^= 1;
  }

  // ---- fused argmax epilogue ----
  // C layout: lanes 0-15 N=l16, VGPR i -> M=i; lanes 16-31 N=l16, VGPR i -> M=8+i
  #pragma unroll
  for (int nf = 0; nf < 2; ++nf) {
    int d = nbase + nf * 16 + l16;
    if (d < D_DOAS) {
      #pragma unroll
      for (int mf = 0; mf < 4; ++mf) {
        #pragma unroll
        for (int i = 0; i < 8; ++i) {
          int m = mbase + mf * 16 + half * 8 + i;
          if (m < M_TOTAL) {
            unsigned long long pk =
                ((unsigned long long)fkey(acc[mf][nf][i]) << 32) | (unsigned int)d;
            atomicMax(&best[m], pk);
          }
        }
      }
    }
  }
}

// ---------------- kernel 4: gather doas[argmax] ----------------
__global__ void srp_gather(const unsigned long long* __restrict__ best,
                           const float* __restrict__ doas, float* __restrict__ out) {
  int m = blockIdx.x * blockDim.x + threadIdx.x;
  if (m >= M_TOTAL) return;
  unsigned int d = (unsigned int)(best[m] & 0xFFFFFFFFu);
  out[3 * m + 0] = doas[3 * d + 0];
  out[3 * m + 1] = doas[3 * d + 1];
  out[3 * m + 2] = doas[3 * d + 2];
}

// ---------------- host ----------------
extern "C" void kernel_launch(void* const* d_in, const int* in_sizes, int n_in,
                              void* d_out, int out_size, void* d_ws, size_t ws_size,
                              hipStream_t stream) {
  const float* XXs  = (const float*)d_in[0];   // (4,250,513,2,36) f32
  const float* taus = (const float*)d_in[1];   // (2562,8) f32
  const float* doas = (const float*)d_in[2];   // (2562,3) f32
  float* out = (float*)d_out;                  // (1000,3) f32

  char* ws = (char*)d_ws;
  _Float16* Xh = (_Float16*)ws;
  _Float16* Wh = (_Float16*)(ws + XH_BYTES);
  unsigned long long* best = (unsigned long long*)(ws + XH_BYTES + WH_BYTES);

  srp_init_best<<<(M_TOTAL + 255) / 256, 256, 0, stream>>>(best);

  {
    size_t total = (size_t)MP * KP_HALF;
    srp_pack_x<<<(unsigned)((total + 255) / 256), 256, 0, stream>>>(XXs, Xh);
  }
  {
    size_t total = (size_t)NP * KP_HALF;
    srp_build_w<<<(unsigned)((total + 255) / 256), 256, 0, stream>>>(taus, Wh);
  }

  dim3 grid(MP / 64, NP / 128);   // 16 x 21
  srp_gemm<<<grid, 128, 0, stream>>>(Xh, Wh, best);

  srp_gather<<<(M_TOTAL + 255) / 256, 256, 0, stream>>>(best, doas, out);
}
